// PyramidAttentionModel_23605140258838
// MI455X (gfx1250) — matrix operand order repaired and checked by reading, not compiled, not run on hardware
//
#include <hip/hip_runtime.h>
#include <hip/hip_bf16.h>
#include <math.h>

// ---------------------------------------------------------------------------
// Types for WMMA fragments (gfx1250, wave32)
// ---------------------------------------------------------------------------
typedef __attribute__((ext_vector_type(16))) __bf16       v16bf;
typedef __attribute__((ext_vector_type(8)))  float        v8f;
typedef __attribute__((ext_vector_type(4)))  unsigned int u32x4;

union Frag {      // 32 bytes = one WMMA bf16 A or B operand per lane
    u32x4 u[2];
    v16bf v;
};

__device__ __forceinline__ unsigned short f2bf(float f) {
    unsigned u = __float_as_uint(f);
    u += 0x7FFFu + ((u >> 16) & 1u);       // round-to-nearest-even
    return (unsigned short)(u >> 16);
}
__device__ __forceinline__ float bf2f(unsigned short h) {
    return __uint_as_float(((unsigned)h) << 16);
}

// gfx1250 async copy: global -> LDS (ASYNCcnt), 16 bytes per lane.
// ldsoff = byte offset within the workgroup LDS allocation (low 32 bits of
// the generic shared-space pointer), gaddr = 64-bit global byte address.
__device__ __forceinline__ void async_g2l_b128(unsigned ldsoff, unsigned long long gaddr) {
    asm volatile("global_load_async_to_lds_b128 %0, %1, off"
                 :: "v"(ldsoff), "v"(gaddr) : "memory");
}
__device__ __forceinline__ void wait_asynccnt0() {
    asm volatile("s_wait_asynccnt 0x0" ::: "memory");
}

// ---------------------------------------------------------------------------
// DCT per-pixel weights: wbar[i] = (1/5) * sum_u scale_u * cos(pi(2i+1)u/128)
// ---------------------------------------------------------------------------
__global__ void init_wbar(float* wbar) {
    int i = threadIdx.x;                   // 64 threads
    float s = 0.f;
    for (int u = 0; u < 5; ++u) {
        float sc = (u == 0) ? 0.125f : 0.17677669529663687f; // sqrt(1/64), sqrt(2/64)
        s += sc * cosf(3.14159265358979323846f * (2.f * i + 1.f) * (float)u / 128.f);
    }
    wbar[i] = s * 0.2f;
}

// ---------------------------------------------------------------------------
// Build bf16 NHWC(Cpad=32) input buffers for the two 3-channel halves of x
// x: (16, 6, 64, 64) fp32 NCHW
// ---------------------------------------------------------------------------
__global__ void prep_xin(const float* __restrict__ x,
                         unsigned short* __restrict__ xg1,
                         unsigned short* __restrict__ xg2) {
    int idx = blockIdx.x * 256 + threadIdx.x;   // 2*16*4096 threads
    int g  = idx >> 16;
    int bp = idx & 65535;                       // b*4096 + p
    const float* src = x + ((size_t)(bp >> 12) * 6 + g * 3) * 4096 + (bp & 4095);
    unsigned short* dst = (g ? xg2 : xg1) + (size_t)bp * 32;
    for (int c = 0; c < 32; ++c)
        dst[c] = (c < 3) ? f2bf(src[(size_t)c * 4096]) : (unsigned short)0;
}

// Append this group's x (3ch + zero pad) after the conv outputs in the
// concat buffers: conv5in channels [64,96), conv7in channels [128,160)
__global__ void fill_x(const float* __restrict__ x, int goff,
                       unsigned short* __restrict__ c5in,
                       unsigned short* __restrict__ c7in) {
    int bp = blockIdx.x * 256 + threadIdx.x;    // 16*4096 threads
    const float* src = x + ((size_t)(bp >> 12) * 6 + goff) * 4096 + (bp & 4095);
    unsigned short* d5 = c5in + (size_t)bp * 96  + 64;
    unsigned short* d7 = c7in + (size_t)bp * 160 + 128;
    for (int c = 0; c < 32; ++c) {
        unsigned short v = (c < 3) ? f2bf(src[(size_t)c * 4096]) : (unsigned short)0;
        d5[c] = v;
        d7[c] = v;
    }
}

// ---------------------------------------------------------------------------
// Pack fp32 OIHW weights -> bf16 [kh*kw][Cout][Cpad]  (zero for c >= Cin)
// ---------------------------------------------------------------------------
__global__ void pack_w(const float* __restrict__ w, unsigned short* __restrict__ wp,
                       int Cout, int Cin, int Cpad, int KS) {
    int idx = blockIdx.x * 256 + threadIdx.x;
    int total = KS * KS * Cout * Cpad;
    if (idx >= total) return;
    int c  = idx % Cpad;
    int o  = (idx / Cpad) % Cout;
    int kk = idx / (Cpad * Cout);
    int kh = kk / KS, kw = kk % KS;
    float v = (c < Cin) ? w[(((size_t)o * Cin + c) * KS + kh) * KS + kw] : 0.f;
    wp[idx] = f2bf(v);
}

// ---------------------------------------------------------------------------
// Implicit-GEMM conv via v_wmma_f32_16x16x32_bf16.
//   in    : bf16 NHWC-ish, [16*4096][CINP]  (pad channels zeroed)
//   wpack : bf16 [KS*KS][Cout][CINP]        (pad channels zeroed)
//   out   : bf16 [16*4096][outCpad], conv output at channels [0, Cout)
// Block = 256 threads = 8 waves; WG tile = 64(M) x 128(N).
// Wave tile = 32x32 = four 16x16 WMMA accumulators (2 M-frags x 2 N-frags).
// LDS staged via gfx1250 global_load_async_to_lds_b128 (ASYNCcnt).
// ---------------------------------------------------------------------------
template<int CINP, int KS>
__global__ void __launch_bounds__(256)
conv_wmma(const unsigned short* __restrict__ in,
          const unsigned short* __restrict__ wpack,
          const float* __restrict__ bias,
          unsigned short* __restrict__ out,
          int Cout, int outCpad) {
    constexpr int PAD   = KS / 2;
    constexpr int NPIX  = 128;                    // WG pixel tile
    __shared__ __align__(16) unsigned short sW[64 * CINP];    // weight slice
    __shared__ __align__(16) unsigned short sA[NPIX * CINP];  // im2col slice

    const int tid  = threadIdx.x;
    const int lane = tid & 31;
    const int wave = tid >> 5;
    const int mi   = wave >> 2;          // 0..1 : 32-row M block
    const int ni   = wave & 3;           // 0..3 : 32-col N block
    const int half = lane >> 4;
    const int l16  = lane & 15;

    const int p0    = blockIdx.x * NPIX; // 128 consecutive pixels, same batch
    const int b     = p0 >> 12;
    const int mbase = blockIdx.y * 64;

    v8f acc00 = {}, acc01 = {}, acc10 = {}, acc11 = {};

    for (int kh = 0; kh < KS; ++kh) {
        for (int kw = 0; kw < KS; ++kw) {
            const unsigned short* wsrc =
                wpack + ((size_t)(kh * KS + kw) * Cout + mbase) * CINP;

            // gfx1250 speculative prefetch of the next weight slice
            if (kh * KS + kw + 1 < KS * KS)
                __builtin_prefetch(wsrc + (size_t)Cout * CINP, 0, 0);

            // Stage weight slice 64 x CINP via async b128 copies
            for (int i = tid; i < 64 * CINP / 8; i += 256)
                async_g2l_b128((unsigned)(size_t)(&sW[i * 8]),
                               (unsigned long long)(size_t)(&wsrc[(size_t)i * 8]));

            // Stage activation slice 128 x CINP (zero when outside image)
            for (int i = tid; i < NPIX * CINP / 8; i += 256) {
                const int pix = i / (CINP / 8);
                const int kc  = (i % (CINP / 8)) * 8;
                const int pp  = p0 + pix;
                const int ih  = ((pp >> 6) & 63) + kh - PAD;
                const int iw  = (pp & 63) + kw - PAD;
                if (((unsigned)ih < 64u) && ((unsigned)iw < 64u)) {
                    async_g2l_b128(
                        (unsigned)(size_t)(&sA[i * 8]),
                        (unsigned long long)(size_t)(
                            &in[((size_t)(b * 64 + ih) * 64 + iw) * CINP + kc]));
                } else {
                    u32x4 z = {};
                    *(u32x4*)(&sA[i * 8]) = z;     // ds_store_b128, DScnt
                }
            }
            wait_asynccnt0();
            __syncthreads();

            // GEMM slice: K = CINP in steps of 32; 4 WMMA per step
#pragma unroll
            for (int k0 = 0; k0 < CINP; k0 += 32) {
                Frag a0, a1, b0, b1;
                // A (16x32 bf16): lane l16 = row, half selects K-phase.
                const unsigned short* ap0 = &sW[(mi * 32 + l16) * CINP + k0 + 8 * half];
                a0.u[0] = *(const u32x4*)(ap0);
                a0.u[1] = *(const u32x4*)(ap0 + 16);
                const unsigned short* ap1 = ap0 + 16 * CINP;
                a1.u[0] = *(const u32x4*)(ap1);
                a1.u[1] = *(const u32x4*)(ap1 + 16);
                // B (32x16 bf16): lane l16 = col, half selects K-phase.
                const unsigned short* bp0 = &sA[(ni * 32 + l16) * CINP + k0 + 8 * half];
                b0.u[0] = *(const u32x4*)(bp0);
                b0.u[1] = *(const u32x4*)(bp0 + 16);
                const unsigned short* bp1 = bp0 + 16 * CINP;
                b1.u[0] = *(const u32x4*)(bp1);
                b1.u[1] = *(const u32x4*)(bp1 + 16);

                acc00 = __builtin_amdgcn_wmma_f32_16x16x32_bf16(
                            false, a0.v, false, b0.v, (short)0, acc00, false, false);
                acc01 = __builtin_amdgcn_wmma_f32_16x16x32_bf16(
                            false, a0.v, false, b1.v, (short)0, acc01, false, false);
                acc10 = __builtin_amdgcn_wmma_f32_16x16x32_bf16(
                            false, a1.v, false, b0.v, (short)0, acc10, false, false);
                acc11 = __builtin_amdgcn_wmma_f32_16x16x32_bf16(
                            false, a1.v, false, b1.v, (short)0, acc11, false, false);
            }
            __syncthreads();
        }
    }

    // Epilogue: bias + ReLU + bf16 pack; lane's 8 rows are 8 contiguous channels
    v8f accs[2][2] = {{acc00, acc01}, {acc10, acc11}};
#pragma unroll
    for (int am = 0; am < 2; ++am) {
#pragma unroll
        for (int bn = 0; bn < 2; ++bn) {
            const int cbase = mbase + mi * 32 + am * 16 + 8 * half;
            const int p     = p0 + ni * 32 + bn * 16 + l16;
            unsigned short o16[8];
#pragma unroll
            for (int r = 0; r < 8; ++r) {
                float v = accs[am][bn][r] + bias[cbase + r];
                o16[r] = f2bf(v > 0.f ? v : 0.f);
            }
            u32x4 st;
#pragma unroll
            for (int i = 0; i < 4; ++i)
                st[i] = (unsigned)o16[2 * i] | ((unsigned)o16[2 * i + 1] << 16);
            *(u32x4*)(&out[(size_t)p * outCpad + cbase]) = st;
        }
    }
}

// ---------------------------------------------------------------------------
// Per-branch reduction: gap mean + DCT weighted sum for 448 channels.
// a3 lives in c5in[.., 0:64), a5 in c7in[.., 0:128), a7 in a7buf[.., 0:256).
// One block per (b, c_local).
// ---------------------------------------------------------------------------
__global__ void reduce_branch(const unsigned short* __restrict__ c5in,
                              const unsigned short* __restrict__ c7in,
                              const unsigned short* __restrict__ a7buf,
                              const float* __restrict__ wbar,
                              float* __restrict__ gap, float* __restrict__ dct,
                              int chan_base) {
    const int b = blockIdx.x / 448;
    const int c = blockIdx.x % 448;
    const unsigned short* src;
    int cp, cc;
    if (c < 64)       { src = c5in;  cp = 96;  cc = c;       }
    else if (c < 192) { src = c7in;  cp = 160; cc = c - 64;  }
    else              { src = a7buf; cp = 256; cc = c - 192; }

    float s = 0.f, sw = 0.f;
    for (int p = threadIdx.x; p < 4096; p += 256) {
        float v = bf2f(src[((size_t)(b * 4096 + p)) * cp + cc]);
        s  += v;
        sw += v * wbar[p >> 6] * wbar[p & 63];
    }
    __shared__ float ss[256], sd[256];
    ss[threadIdx.x] = s; sd[threadIdx.x] = sw;
    __syncthreads();
    for (int st = 128; st > 0; st >>= 1) {
        if (threadIdx.x < st) {
            ss[threadIdx.x] += ss[threadIdx.x + st];
            sd[threadIdx.x] += sd[threadIdx.x + st];
        }
        __syncthreads();
    }
    if (threadIdx.x == 0) {
        gap[b * 1792 + chan_base + c] = ss[0] * (1.f / 4096.f);
        dct[b * 1792 + chan_base + c] = sd[0];
    }
}

// ---------------------------------------------------------------------------
// Heads: gap_pred = gap @ gap_w.T + gap_b ; dct_pred = BN(dct) @ dct_w.T + dct_b
// One wave32 per (kind, b, j) dot product. 384 tasks.
// ---------------------------------------------------------------------------
__global__ void head(const float* __restrict__ gap, const float* __restrict__ dct,
                     const float* __restrict__ gap_w, const float* __restrict__ gap_b,
                     const float* __restrict__ dct_w, const float* __restrict__ dct_b,
                     const float* __restrict__ gamma, const float* __restrict__ beta,
                     const float* __restrict__ mean,  const float* __restrict__ var,
                     float* __restrict__ out) {
    int task = blockIdx.x * 8 + (threadIdx.x >> 5);
    if (task >= 384) return;
    int lane = threadIdx.x & 31;
    int kind = task / 192;
    int rem  = task % 192;
    int b = rem / 12, j = rem % 12;
    float s = 0.f;
    if (kind == 0) {
        for (int k = lane; k < 1792; k += 32)
            s += gap[b * 1792 + k] * gap_w[j * 1792 + k];
    } else {
        for (int k = lane; k < 1792; k += 32) {
            float v = gamma[k] * (dct[b * 1792 + k] - mean[k]) * rsqrtf(var[k] + 1e-5f) + beta[k];
            s += v * dct_w[j * 1792 + k];
        }
    }
    for (int o = 16; o; o >>= 1) s += __shfl_xor(s, o, 32);
    if (lane == 0)
        out[kind * 192 + b * 12 + j] = s + (kind == 0 ? gap_b[j] : dct_b[j]);
}

// ---------------------------------------------------------------------------
extern "C" void kernel_launch(void* const* d_in, const int* in_sizes, int n_in,
                              void* d_out, int out_size, void* d_ws, size_t ws_size,
                              hipStream_t stream) {
    (void)in_sizes; (void)n_in; (void)out_size; (void)ws_size;
    const float* x = (const float*)d_in[0];
    const float* gap_w = (const float*)d_in[25];
    const float* gap_b = (const float*)d_in[26];
    const float* dct_w = (const float*)d_in[27];
    const float* dct_b = (const float*)d_in[28];
    const float* bn_g  = (const float*)d_in[29];
    const float* bn_be = (const float*)d_in[30];
    const float* bn_m  = (const float*)d_in[31];
    const float* bn_v  = (const float*)d_in[32];

    char* ws = (char*)d_ws;
    size_t off = 0;
    auto alloc = [&](size_t bytes) -> void* {
        off = (off + 255) & ~(size_t)255;
        void* p = ws + off;
        off += bytes;
        return p;
    };

    float*          wbar  = (float*)alloc(64 * sizeof(float));
    float*          gap   = (float*)alloc((size_t)16 * 1792 * sizeof(float));
    float*          dct   = (float*)alloc((size_t)16 * 1792 * sizeof(float));
    unsigned short* xin1  = (unsigned short*)alloc((size_t)16 * 4096 * 32  * 2);
    unsigned short* xin2  = (unsigned short*)alloc((size_t)16 * 4096 * 32  * 2);
    unsigned short* c5in  = (unsigned short*)alloc((size_t)16 * 4096 * 96  * 2);
    unsigned short* c7in  = (unsigned short*)alloc((size_t)16 * 4096 * 160 * 2);
    unsigned short* a7b   = (unsigned short*)alloc((size_t)16 * 4096 * 256 * 2);
    unsigned short* wp3   = (unsigned short*)alloc((size_t)9  * 64  * 32  * 2);
    unsigned short* wp5   = (unsigned short*)alloc((size_t)25 * 128 * 96  * 2);
    unsigned short* wp7   = (unsigned short*)alloc((size_t)49 * 256 * 160 * 2);

    init_wbar<<<1, 64, 0, stream>>>(wbar);
    prep_xin<<<512, 256, 0, stream>>>(x, xin1, xin2);

    for (int br = 0; br < 4; ++br) {
        const int g    = br >> 1;            // 0: channels 0-2, 1: channels 3-5
        const int widx = 1 + br * 6;
        const float* c3w = (const float*)d_in[widx + 0];
        const float* c3b = (const float*)d_in[widx + 1];
        const float* c5w = (const float*)d_in[widx + 2];
        const float* c5b = (const float*)d_in[widx + 3];
        const float* c7w = (const float*)d_in[widx + 4];
        const float* c7b = (const float*)d_in[widx + 5];
        unsigned short* xin = g ? xin2 : xin1;

        if ((br & 1) == 0)   // x channels in concat buffers change per group
            fill_x<<<256, 256, 0, stream>>>(x, g * 3, c5in, c7in);

        pack_w<<<(9  * 64  * 32  + 255) / 256, 256, 0, stream>>>(c3w, wp3, 64, 3,   32,  3);
        pack_w<<<(25 * 128 * 96  + 255) / 256, 256, 0, stream>>>(c5w, wp5, 128, 67, 96,  5);
        pack_w<<<(49 * 256 * 160 + 255) / 256, 256, 0, stream>>>(c7w, wp7, 256, 131, 160, 7);

        conv_wmma<32, 3><<<dim3(512, 1), 256, 0, stream>>>(xin,  wp3, c3b, c5in, 64,  96);
        conv_wmma<96, 5><<<dim3(512, 2), 256, 0, stream>>>(c5in, wp5, c5b, c7in, 128, 160);
        conv_wmma<160, 7><<<dim3(512, 4), 256, 0, stream>>>(c7in, wp7, c7b, a7b, 256, 256);

        reduce_branch<<<16 * 448, 256, 0, stream>>>(c5in, c7in, a7b, wbar,
                                                    gap, dct, br * 448);
    }

    head<<<48, 256, 0, stream>>>(gap, dct, gap_w, gap_b, dct_w, dct_b,
                                 bn_g, bn_be, bn_m, bn_v, (float*)d_out);
}
